// FasterBERTSelfAttention_25417616458433
// MI455X (gfx1250) — matrix operand order repaired and checked by reading, not compiled, not run on hardware
//
#include <hip/hip_runtime.h>
#include <hip/hip_bf16.h>

typedef __attribute__((ext_vector_type(16))) __bf16 v16bf;
typedef __attribute__((ext_vector_type(8)))  float  v8f;
typedef __attribute__((ext_vector_type(4)))  unsigned int u32x4;
typedef __attribute__((ext_vector_type(8)))  int i32x8;
typedef __attribute__((ext_vector_type(4)))  int i32x4;

#define NB   2
#define SEQ  2048
#define HIDN 1024
#define NHD  16
#define HD   64

#if __has_builtin(__builtin_amdgcn_tensor_load_to_lds) && __has_builtin(__builtin_amdgcn_s_wait_tensorcnt)
#define USE_TDM 1
#else
#define USE_TDM 0
#endif

union BF16x16 { unsigned short u[16]; uint4 q[2]; v16bf v; };

// native f32 -> bf16 convert (RNE); pairs can fuse into v_cvt_pk_bf16_f32
__device__ __forceinline__ unsigned short f2bf(float f) {
    union { __bf16 b; unsigned short s; } u;
    u.b = (__bf16)f;
    return u.s;
}

#if USE_TDM
// 1-D contiguous bf16 tile move, global -> LDS, via the Tensor Data Mover.
// nelem <= 65535 (tile_dim0 is 16 bits); here always 4096 (8 KB).
__device__ __forceinline__ void tdm_load_bf16_lin(void* lds_dst, const void* gsrc,
                                                  unsigned nelem) {
    unsigned long long ga = (unsigned long long)gsrc;
    unsigned lds = (unsigned)(unsigned long long)lds_dst;
    u32x4 g0 = { 1u,                                   // count=1 valid user D#
                 lds,                                  // lds_addr
                 (unsigned)ga,                         // global_addr[31:0]
                 (unsigned)((ga >> 32) & 0x01FFFFFFu)  // global_addr[56:32]
                     | (2u << 30) };                   // type=2 (image)
    i32x8 g1 = { (int)(1u << 16),       // data_size=1 -> 2 bytes/elem
                 (int)(nelem << 16),    // tensor_dim0[15:0]
                 (int)(1u << 16),       // tensor_dim0[31:16]=0, tensor_dim1=1
                 (int)(nelem << 16),    // tile_dim0
                 1,                     // tile_dim1=1
                 (int)nelem,            // tensor_dim0_stride lo
                 0, 0 };
    i32x4 z4 = { 0, 0, 0, 0 };
    i32x8 z8 = { 0, 0, 0, 0, 0, 0, 0, 0 };
    __builtin_amdgcn_tensor_load_to_lds(g0, g1, z4, z4, z8, 0);
}
#endif

// ---------------------------------------------------------------------------
// Kernel 1: Y = hs @ W + bias, then interleaved RoPE, store bf16 [B,NH,S,HD]
// Block: 256 threads (8 waves). Tile: 128 (M) x 64 (N == one head). K step 32.
// W tile is stored TRANSPOSED in LDS so B-fragments are contiguous b128 loads.
// ---------------------------------------------------------------------------
__global__ __launch_bounds__(256) void qkv_rope_kernel(
    const float* __restrict__ hs, const float* __restrict__ pe,
    const float* __restrict__ W,  const float* __restrict__ bias,
    unsigned short* __restrict__ dst)
{
    __shared__ union {
        struct { unsigned short A[128 * 32]; unsigned short BwT[64 * 32]; } t;
        float Y[128 * 64];
    } sm;

    const int tid  = threadIdx.x;
    const int lane = tid & 31;
    const int wave = tid >> 5;
    const int half = lane >> 4;
    const int l16  = lane & 15;
    const int row0 = blockIdx.x * 128;   // global M base (= b*SEQ + s)
    const int col0 = blockIdx.y * 64;    // global N base (= head*64)

    v8f acc[4];
#pragma unroll
    for (int nt = 0; nt < 4; ++nt)
#pragma unroll
        for (int i = 0; i < 8; ++i) acc[nt][i] = 0.f;

    for (int k0 = 0; k0 < HIDN; k0 += 32) {
        if (k0 + 32 < HIDN) {   // prefetch next A/B tiles
            __builtin_prefetch(hs + (size_t)(row0 + (tid >> 1)) * HIDN + k0 + 32, 0, 3);
            __builtin_prefetch(W + (size_t)(k0 + 32 + (tid >> 3)) * HIDN + col0, 0, 3);
        }
        __syncthreads();
        {   // A tile: 128x32 fp32 -> bf16 LDS (row-major); 16 elements / thread
            int base = tid * 16;
            int r = base >> 5, c = base & 31;
            const float4* src4 = (const float4*)(hs + (size_t)(row0 + r) * HIDN + k0 + c);
            unsigned short* dA = sm.t.A + r * 32 + c;
#pragma unroll
            for (int jj = 0; jj < 4; ++jj) {
                float4 t = src4[jj];
                dA[jj * 4 + 0] = f2bf(t.x); dA[jj * 4 + 1] = f2bf(t.y);
                dA[jj * 4 + 2] = f2bf(t.z); dA[jj * 4 + 3] = f2bf(t.w);
            }
        }
        {   // B tile: 32(k) x 64(n) fp32 -> bf16 LDS TRANSPOSED [n][k]
            int base = tid * 8;
            int r = base >> 6;     // k row 0..31
            int c = base & 63;     // n col, multiple of 8
            const float4* src4 = (const float4*)(W + (size_t)(k0 + r) * HIDN + col0 + c);
            float4 t0 = src4[0], t1 = src4[1];
            unsigned short* dB = sm.t.BwT + c * 32 + r;   // stride 32 along n
            dB[0 * 32] = f2bf(t0.x); dB[1 * 32] = f2bf(t0.y);
            dB[2 * 32] = f2bf(t0.z); dB[3 * 32] = f2bf(t0.w);
            dB[4 * 32] = f2bf(t1.x); dB[5 * 32] = f2bf(t1.y);
            dB[6 * 32] = f2bf(t1.z); dB[7 * 32] = f2bf(t1.w);
        }
        __syncthreads();

        // A fragment (16x32, this wave's 16 rows)
        BF16x16 af;
        const unsigned short* Arow = sm.t.A + (wave * 16 + l16) * 32;
#pragma unroll
        for (int j = 0; j < 16; ++j) {
            int k = (half << 4) + j;
            af.u[j] = Arow[k];
        }
#pragma unroll
        for (int nt = 0; nt < 4; ++nt) {
            BF16x16 bf;   // contiguous in transposed layout: two ds_load_b128
            const unsigned short* Bp = sm.t.BwT + (nt * 16 + l16) * 32 + half * 16;
            bf.q[0] = *(const uint4*)(Bp);
            bf.q[1] = *(const uint4*)(Bp + 8);
            acc[nt] = __builtin_amdgcn_wmma_f32_16x16x32_bf16(
                false, af.v, false, bf.v, (short)0, acc[nt], false, false);
        }
    }

    __syncthreads();
    // epilogue: + bias, stage fp32 tile in LDS (RoPE pairs columns d and d^1)
#pragma unroll
    for (int nt = 0; nt < 4; ++nt) {
        int d = nt * 16 + l16;
        float bv = bias[col0 + d];
#pragma unroll
        for (int i = 0; i < 8; ++i)
            sm.Y[(wave * 16 + i + half * 8) * 64 + d] = acc[nt][i] + bv;
    }
    __syncthreads();
    {   // RoPE + bf16 store: 32 elements / thread
        int r  = tid >> 1;
        int c0 = (tid & 1) << 5;
        int grow = row0 + r;
        int bidx = grow >> 11;        // / SEQ
        int s    = grow & (SEQ - 1);
        const float* pes  = pe + s * HD;        // [0..31]=sin, [32..63]=cos halves
        const float* Yrow = sm.Y + r * 64;
        unsigned short* op = dst + (((size_t)bidx * NHD + blockIdx.y) * SEQ + s) * HD;
#pragma unroll
        for (int j = 0; j < 32; ++j) {
            int d = c0 + j;
            float cs = pes[32 + (d >> 1)];
            float sn = pes[d >> 1];
            float x  = Yrow[d];
            float xp = Yrow[d ^ 1];
            float rot = (d & 1) ? xp : -xp;
            op[d] = f2bf(x * cs + rot * sn);
        }
    }
}

// ---------------------------------------------------------------------------
// Kernel 2: flash attention over bf16 q/k/v [B,NH,S,HD].
// Block: 128 threads (4 waves), 64 q-rows per block, 64-key tiles.
// K tiles are double-buffered via TENSOR_LOAD_TO_LDS: the DMA for tile kt+1
// is issued before computing tile kt, and completion is enforced with
// s_wait_tensorcnt(1) (TENSORcnt completes in order).
// V is staged transposed [d][kpos] so PV B-fragments are b128 loads.
// ---------------------------------------------------------------------------
__global__ __launch_bounds__(128) void attn_kernel(
    const unsigned short* __restrict__ qws,
    const unsigned short* __restrict__ kws,
    const unsigned short* __restrict__ vws,
    const float* __restrict__ mask,
    float* __restrict__ out)
{
    __shared__ unsigned short Ksh[2][64 * 64];  // [buf][kpos][d]
    __shared__ unsigned short VshT[64 * 64];    // [d][kpos]
    __shared__ unsigned short Psh[64 * 64];     // Q staging, then per-wave P tiles

    const int tid  = threadIdx.x;
    const int lane = tid & 31;
    const int wave = tid >> 5;     // 0..3
    const int half = lane >> 4;
    const int l16  = lane & 15;
    const int h    = blockIdx.y;
    const int b    = blockIdx.z;
    const int q0   = blockIdx.x * 64;
    const size_t headoff = ((size_t)b * NHD + h) * SEQ * HD;
    const int NT = SEQ / 64;

#if USE_TDM
    if (wave == 0) {
        tdm_load_bf16_lin(Psh, qws + headoff + (size_t)q0 * HD, 64 * HD);
        tdm_load_bf16_lin(Ksh[0], kws + headoff, 64 * HD);   // K tile 0
        __builtin_amdgcn_s_wait_tensorcnt(0);
    }
#else
    {   // stage Q tile (64x64 bf16 = 512 uint4)
        const uint4* src = (const uint4*)(qws + headoff + (size_t)q0 * HD);
        uint4* d4 = (uint4*)Psh;
#pragma unroll
        for (int j = 0; j < 4; ++j) d4[tid + j * 128] = src[tid + j * 128];
    }
#endif
    __syncthreads();

    // Q A-fragments (this wave's 16 rows; K = d, two 16x32 frags), held in VGPRs
    BF16x16 aQ[2];
    {
        const unsigned short* Qrow = Psh + (wave * 16 + l16) * 64;
#pragma unroll
        for (int g = 0; g < 2; ++g)
#pragma unroll
            for (int j = 0; j < 16; ++j) {
                int k = g * 32 + (half << 4) + j;
                aQ[g].u[j] = Qrow[k];
            }
    }

    float mrow[8], lrow[8];
    v8f O[4];
#pragma unroll
    for (int i = 0; i < 8; ++i) { mrow[i] = -1e30f; lrow[i] = 0.f; }
#pragma unroll
    for (int nt = 0; nt < 4; ++nt)
#pragma unroll
        for (int i = 0; i < 8; ++i) O[nt][i] = 0.f;

    for (int kt = 0; kt < NT; ++kt) {
        const int cur = kt & 1;
        if (kt + 1 < NT)
            __builtin_prefetch(vws + headoff + (size_t)(kt + 1) * 64 * HD + tid * 32, 0, 3);
        __syncthreads();   // all waves done reading previous K buf / V tile

#if USE_TDM
        if (wave == 0 && kt + 1 < NT)   // async DMA of NEXT K tile into other buf
            tdm_load_bf16_lin(Ksh[cur ^ 1],
                              kws + headoff + (size_t)(kt + 1) * 64 * HD, 64 * HD);
#else
        {   // fallback: copy current K tile
            const uint4* ks = (const uint4*)(kws + headoff + (size_t)kt * 64 * HD);
            uint4* kd = (uint4*)Ksh[cur];
#pragma unroll
            for (int j = 0; j < 4; ++j) kd[tid + j * 128] = ks[tid + j * 128];
        }
#endif
        {   // V tile -> LDS transposed [d][kpos] (overlaps with K DMA)
            const uint4* vs = (const uint4*)(vws + headoff + (size_t)kt * 64 * HD);
#pragma unroll
            for (int j = 0; j < 4; ++j) {
                uint4 t = vs[tid + j * 128];
                int linear = (tid + j * 128) * 8;
                int r = linear >> 6;      // kpos
                int c = linear & 63;      // d base (multiple of 8)
                const unsigned short* p = (const unsigned short*)&t;
#pragma unroll
                for (int e = 0; e < 8; ++e) VshT[(c + e) * 64 + r] = p[e];
            }
        }
#if USE_TDM
        if (wave == 0) {   // tile kt is done when <=1 DMA (kt+1) is outstanding
            if (kt + 1 < NT) __builtin_amdgcn_s_wait_tensorcnt(1);
            else             __builtin_amdgcn_s_wait_tensorcnt(0);
        }
#endif
        __syncthreads();

        // scores = Q @ K^T  (B-frag: kdim = d contiguous in Ksh row)
        v8f sc[4];
#pragma unroll
        for (int nt = 0; nt < 4; ++nt)
#pragma unroll
            for (int i = 0; i < 8; ++i) sc[nt][i] = 0.f;
#pragma unroll
        for (int kk = 0; kk < 2; ++kk)
#pragma unroll
            for (int nt = 0; nt < 4; ++nt) {
                BF16x16 bk;
                const unsigned short* Krow =
                    Ksh[cur] + (nt * 16 + l16) * 64 + kk * 32 + half * 16;
                bk.q[0] = *(const uint4*)(Krow);
                bk.q[1] = *(const uint4*)(Krow + 8);
                sc[nt] = __builtin_amdgcn_wmma_f32_16x16x32_bf16(
                    false, aQ[kk].v, false, bk.v, (short)0, sc[nt], false, false);
            }

        // scale + additive mask
#pragma unroll
        for (int nt = 0; nt < 4; ++nt) {
            float madd = mask[b * SEQ + kt * 64 + nt * 16 + l16] * -1e8f;
#pragma unroll
            for (int i = 0; i < 8; ++i) sc[nt][i] = sc[nt][i] * 0.125f + madd;
        }

        // online softmax (a row's 16 columns live across lanes 0-15 / 16-31)
#pragma unroll
        for (int i = 0; i < 8; ++i) {
            float mt = fmaxf(fmaxf(sc[0][i], sc[1][i]), fmaxf(sc[2][i], sc[3][i]));
#pragma unroll
            for (int off = 1; off < 16; off <<= 1)
                mt = fmaxf(mt, __shfl_xor(mt, off, 32));
            float mn = fmaxf(mrow[i], mt);
            float alpha = __expf(mrow[i] - mn);
            mrow[i] = mn;
            float psum = 0.f;
#pragma unroll
            for (int nt = 0; nt < 4; ++nt) {
                float p = __expf(sc[nt][i] - mn);
                sc[nt][i] = p;
                psum += p;
            }
#pragma unroll
            for (int off = 1; off < 16; off <<= 1)
                psum += __shfl_xor(psum, off, 32);
            lrow[i] = lrow[i] * alpha + psum;
#pragma unroll
            for (int nt = 0; nt < 4; ++nt) O[nt][i] *= alpha;
        }

        // D-layout -> A-layout reshape of probs through per-wave LDS region
        unsigned short* PW = Psh + wave * 16 * 64;
#pragma unroll
        for (int nt = 0; nt < 4; ++nt)
#pragma unroll
            for (int i = 0; i < 8; ++i)
                PW[(i + half * 8) * 64 + nt * 16 + l16] = f2bf(sc[nt][i]);

        // O += P @ V
#pragma unroll
        for (int kk = 0; kk < 2; ++kk) {
            BF16x16 aP;
            const unsigned short* Prow = PW + l16 * 64 + kk * 32;
#pragma unroll
            for (int j = 0; j < 16; ++j) {
                int k = (half << 4) + j;
                aP.u[j] = Prow[k];
            }
#pragma unroll
            for (int nt = 0; nt < 4; ++nt) {
                BF16x16 bv;   // contiguous in transposed V: two ds_load_b128
                const unsigned short* Vp =
                    VshT + (nt * 16 + l16) * 64 + kk * 32 + half * 16;
                bv.q[0] = *(const uint4*)(Vp);
                bv.q[1] = *(const uint4*)(Vp + 8);
                O[nt] = __builtin_amdgcn_wmma_f32_16x16x32_bf16(
                    false, aP.v, false, bv.v, (short)0, O[nt], false, false);
            }
        }
    }

    // normalize and write [B,S,NH*HD] fp32
#pragma unroll
    for (int nt = 0; nt < 4; ++nt)
#pragma unroll
        for (int i = 0; i < 8; ++i) {
            int r = q0 + wave * 16 + i + half * 8;
            int d = nt * 16 + l16;
            out[((size_t)b * SEQ + r) * HIDN + h * HD + d] = O[nt][i] / lrow[i];
        }
}

// ---------------------------------------------------------------------------
extern "C" void kernel_launch(void* const* d_in, const int* in_sizes, int n_in,
                              void* d_out, int out_size, void* d_ws, size_t ws_size,
                              hipStream_t stream) {
    const float* hs   = (const float*)d_in[0];
    const float* mask = (const float*)d_in[1];
    const float* pe   = (const float*)d_in[2];
    const float* Wq   = (const float*)d_in[3];
    const float* bq   = (const float*)d_in[4];
    const float* Wk   = (const float*)d_in[5];
    const float* bk   = (const float*)d_in[6];
    const float* Wv   = (const float*)d_in[7];
    const float* bv   = (const float*)d_in[8];
    float* outp = (float*)d_out;

    const size_t per = (size_t)NB * NHD * SEQ * HD;   // bf16 elements per tensor
    unsigned short* qws = (unsigned short*)d_ws;
    unsigned short* kws = qws + per;
    unsigned short* vws = kws + per;

    dim3 g1(NB * SEQ / 128, HIDN / 64, 1);
    qkv_rope_kernel<<<g1, 256, 0, stream>>>(hs, pe, Wq, bq, qws);
    qkv_rope_kernel<<<g1, 256, 0, stream>>>(hs, pe, Wk, bk, kws);
    qkv_rope_kernel<<<g1, 256, 0, stream>>>(hs, pe, Wv, bv, vws);

    dim3 g2(SEQ / 64, NHD, NB);
    attn_kernel<<<g2, 128, 0, stream>>>(qws, kws, vws, mask, outp);
}
